// BasicAttentionModel_40879498728927
// MI455X (gfx1250) — matrix operand order repaired and checked by reading, not compiled, not run on hardware
//
#include <hip/hip_runtime.h>
#include <hip/hip_bf16.h>
#include <math.h>

// ---------------------------------------------------------------------------
// Types for CDNA5 WMMA (wave32, 16x16x32 bf16 -> f32 accum) and TDM
// ---------------------------------------------------------------------------
typedef __attribute__((ext_vector_type(16))) __bf16 v16bf;
typedef __attribute__((ext_vector_type(8)))  __bf16 v8bf;
typedef __attribute__((ext_vector_type(8)))  float  v8f;
typedef __attribute__((ext_vector_type(4)))  unsigned int u32x4;
typedef __attribute__((ext_vector_type(8)))  int i32x8;
typedef __attribute__((ext_vector_type(4)))  int i32x4;

#define GAT_HEADS 3
#define NNODES 50000
#define NEDGES 800000

// TDM builtin arity differs between toolchains; the gfx1250 TDM header ships
// only with the 6-arg (therock) toolchain -> use it as a compile-time probe.
#if defined(__has_builtin)
#if __has_builtin(__builtin_amdgcn_tensor_load_to_lds)
#define HAVE_TDM 1
#endif
#endif
#if __has_include(<hip/amd_detail/amd_gfx1250_TDM.h>)
#define TDM_SIX_ARGS 1
#endif

static __device__ __forceinline__ __bf16 f2bf(float f) { return (__bf16)f; }
static __device__ __forceinline__ float leakyf(float v, float a) { return v > 0.f ? v : a * v; }

static __device__ __forceinline__ v8f wmma_bf16(v16bf a, v16bf b, v8f c) {
  // 8 args: (neg_a, A, neg_b, B, c_mod, C, reuse_a, reuse_b)
  return __builtin_amdgcn_wmma_f32_16x16x32_bf16(false, a, false, b, (short)0, c, false, false);
}

// order-preserving float<->int encode for atomicMax on floats
static __device__ __forceinline__ int f2ord(float f) {
  int i = __float_as_int(f);
  return (i >= 0) ? i : (i ^ 0x7fffffff);
}
static __device__ __forceinline__ float ord2f(int i) {
  return __int_as_float((i >= 0) ? i : (i ^ 0x7fffffff));
}

// ---------------------------------------------------------------------------
// BatchNorm (training-mode, biased var): column stats + apply
// ---------------------------------------------------------------------------
__global__ __launch_bounds__(256) void colstats_kernel(const float* __restrict__ src,
                                                       int rows, int cols,
                                                       float* __restrict__ sums,
                                                       float* __restrict__ sqs) {
  __shared__ float ls[32], lq[32];
  if (threadIdx.x < 32) { ls[threadIdx.x] = 0.f; lq[threadIdx.x] = 0.f; }
  __syncthreads();
  for (int r = blockIdx.x * blockDim.x + threadIdx.x; r < rows; r += gridDim.x * blockDim.x) {
    const float* p = src + (size_t)r * cols;
    for (int c = 0; c < cols; ++c) {
      float v = p[c];
      atomicAdd(&ls[c], v);
      atomicAdd(&lq[c], v * v);
    }
  }
  __syncthreads();
  if ((int)threadIdx.x < cols) {
    atomicAdd(&sums[threadIdx.x], ls[threadIdx.x]);
    atomicAdd(&sqs[threadIdx.x],  lq[threadIdx.x]);
  }
}

__global__ __launch_bounds__(256) void bn_apply_kernel(const float* __restrict__ src,
                                                       float* __restrict__ dst,
                                                       const float* __restrict__ sums,
                                                       const float* __restrict__ sqs,
                                                       const float* __restrict__ g,
                                                       const float* __restrict__ b,
                                                       int rows, int cols) {
  int idx = blockIdx.x * blockDim.x + threadIdx.x;
  if (idx >= rows * cols) return;
  int c = idx % cols;
  float inv_rows = 1.f / (float)rows;
  float mean = sums[c] * inv_rows;
  float var  = sqs[c] * inv_rows - mean * mean;
  dst[idx] = (src[idx] - mean) * rsqrtf(var + 1e-5f) * g[c] + b[c];
}

// ---------------------------------------------------------------------------
// Generic small-K GEMM: C[rows,N] = A[rows,K] @ W[K,N], bf16 WMMA, f32 accum.
// One wave computes one 16x16 output tile; K <= 32 (zero-padded into frags).
// rows and N are multiples of 16.
// ---------------------------------------------------------------------------
__global__ __launch_bounds__(256) void gemm16_bf16_kernel(const float* __restrict__ A,
                                                          const float* __restrict__ W,
                                                          float* __restrict__ C,
                                                          int rows, int K, int N) {
  int wave = threadIdx.x >> 5;
  int lane = threadIdx.x & 31;
  int nTiles = N >> 4;
  int tile = blockIdx.x * 8 + wave;
  int totalTiles = (rows >> 4) * nTiles;
  if (tile >= totalTiles) return;            // wave-uniform: EXEC stays all-1
  int rowTile = tile / nTiles;
  int nTile   = tile % nTiles;
  int m    = lane & 15;
  int kgrp = lane >> 4;

  // A fragment: 16x32 (K zero-padded); lane holds row m, K runs kgrp*8..+7 and 16+kgrp*8..+7
  const float* arow = A + (size_t)(rowTile * 16 + m) * K;
  v16bf a;
#pragma unroll
  for (int i = 0; i < 8; ++i) {
    int klo = kgrp * 8 + i;
    int khi = 16 + kgrp * 8 + i;
    a[i]     = f2bf(klo < K ? arow[klo] : 0.f);
    a[i + 8] = f2bf(khi < K ? arow[khi] : 0.f);
  }

  // B fragment: 32x16; lane holds col n, K = kgrp*8 + j (j<8), 16 + kgrp*8 + (j-8) (j>=8)
  int n = lane & 15;
  int ncol = nTile * 16 + n;
  v16bf b;
#pragma unroll
  for (int j = 0; j < 16; ++j) {
    int k = (j < 8) ? kgrp * 8 + j : 16 + kgrp * 8 + (j - 8);
    b[j] = f2bf(k < K ? W[(size_t)k * N + ncol] : 0.f);
  }

  v8f c = {};
  c = wmma_bf16(a, b, c);

#pragma unroll
  for (int r = 0; r < 8; ++r) {
    int row = rowTile * 16 + r + 8 * kgrp;
    C[(size_t)row * N + ncol] = c[r];
  }
}

// ---------------------------------------------------------------------------
// GAT attention pieces
// ---------------------------------------------------------------------------
__global__ __launch_bounds__(256) void attn_score_kernel(const float* __restrict__ h,
                                                         const float* __restrict__ a_s,
                                                         const float* __restrict__ a_d,
                                                         float* __restrict__ asrc,
                                                         float* __restrict__ adst,
                                                         int N, int outdim) {
  int idx = blockIdx.x * blockDim.x + threadIdx.x;
  if (idx >= N * GAT_HEADS) return;
  int nid = idx / GAT_HEADS, hh = idx % GAT_HEADS;
  const float* hv = h + ((size_t)nid * GAT_HEADS + hh) * outdim;
  const float* ws = a_s + hh * outdim;
  const float* wd = a_d + hh * outdim;
  float ss = 0.f, sd = 0.f;
  for (int d = 0; d < outdim; ++d) { float v = hv[d]; ss += v * ws[d]; sd += v * wd[d]; }
  asrc[idx] = ss;
  adst[idx] = sd;
}

__global__ __launch_bounds__(256) void init_amax_kernel(int* __restrict__ amax, int n) {
  int idx = blockIdx.x * blockDim.x + threadIdx.x;
  if (idx < n) amax[idx] = f2ord(-INFINITY);
}

__global__ __launch_bounds__(256) void edge_score_kernel(const int* __restrict__ ei,
                                                         int E, int Etot,
                                                         const float* __restrict__ asrc,
                                                         const float* __restrict__ adst,
                                                         float* __restrict__ al,
                                                         int* __restrict__ amax) {
  int idx = blockIdx.x * blockDim.x + threadIdx.x;
  if (idx >= Etot * GAT_HEADS) return;
  int e = idx / GAT_HEADS, hh = idx % GAT_HEADS;
  int s, d;
  if (e < E) { s = ei[e]; d = ei[E + e]; } else { s = d = e - E; }
  float v = asrc[s * GAT_HEADS + hh] + adst[d * GAT_HEADS + hh];
  v = leakyf(v, 0.2f);
  al[idx] = v;
  atomicMax(&amax[d * GAT_HEADS + hh], f2ord(v));
}

__global__ __launch_bounds__(256) void edge_exp_kernel(const int* __restrict__ ei,
                                                       int E, int Etot,
                                                       const float* __restrict__ al,
                                                       const int* __restrict__ amax,
                                                       float* __restrict__ ea,
                                                       float* __restrict__ denom) {
  int idx = blockIdx.x * blockDim.x + threadIdx.x;
  if (idx >= Etot * GAT_HEADS) return;
  int e = idx / GAT_HEADS, hh = idx % GAT_HEADS;
  int d;
  if (e < E) { d = ei[E + e]; } else { d = e - E; }
  float mx = ord2f(amax[d * GAT_HEADS + hh]);
  float v = __expf(al[idx] - mx);
  ea[idx] = v;
  atomicAdd(&denom[d * GAT_HEADS + hh], v);
}

__global__ __launch_bounds__(256) void edge_agg_kernel(const int* __restrict__ ei,
                                                       int E, int Etot,
                                                       const float* __restrict__ ea,
                                                       const float* __restrict__ denom,
                                                       const float* __restrict__ h,
                                                       float* __restrict__ agg,
                                                       int outdim) {
  int idx = blockIdx.x * blockDim.x + threadIdx.x;
  if (idx >= Etot * GAT_HEADS) return;
  int e = idx / GAT_HEADS, hh = idx % GAT_HEADS;
  int s, d;
  if (e < E) { s = ei[e]; d = ei[E + e]; } else { s = d = e - E; }
  float w = ea[idx] / (denom[d * GAT_HEADS + hh] + 1e-16f);
  const float* hs = h + ((size_t)s * GAT_HEADS + hh) * outdim;
  float* ag = agg + ((size_t)d * GAT_HEADS + hh) * outdim;
  for (int dd = 0; dd < outdim; ++dd) atomicAdd(&ag[dd], hs[dd] * w);
}

__global__ __launch_bounds__(256) void head_mean_kernel(const float* __restrict__ agg,
                                                        const float* __restrict__ bias,
                                                        float* __restrict__ xout,
                                                        int N, int outdim) {
  int idx = blockIdx.x * blockDim.x + threadIdx.x;
  if (idx >= N * outdim) return;
  int n = idx / outdim, dd = idx % outdim;
  size_t b0 = (size_t)n * GAT_HEADS * outdim + dd;
  float s = agg[b0] + agg[b0 + outdim] + agg[b0 + 2 * (size_t)outdim];
  xout[idx] = s * (1.f / 3.f) + bias[dd];
}

// ---------------------------------------------------------------------------
// Fused edge-MLP: gather [x3[s] | x3[d] | e] (138 -> pad 160), then five
// chained bf16 WMMA GEMMs per 16-edge tile entirely in LDS/registers.
// Block = 256 threads = 8 waves; wave handles 16 edges -> 128 edges/block.
// mw1 (138x64 f32) is DMA'd into LDS by the Tensor Data Mover.
// ---------------------------------------------------------------------------
__global__ __launch_bounds__(256) void edge_mlp_kernel(const float* __restrict__ x3,
                                                       const int* __restrict__ ei,
                                                       const float* __restrict__ en,
                                                       const float* __restrict__ mw1, const float* __restrict__ mb1,
                                                       const float* __restrict__ mw2, const float* __restrict__ mb2,
                                                       const float* __restrict__ mw3, const float* __restrict__ mb3,
                                                       const float* __restrict__ mw4, const float* __restrict__ mb4,
                                                       const float* __restrict__ mw5, const float* __restrict__ mb5,
                                                       float* __restrict__ out, int E) {
  __shared__ __bf16 W1s[160 * 64];   // K padded 138->160
  __shared__ __bf16 W2s[64 * 32];
  __shared__ __bf16 W3s[32 * 16];
  __shared__ __bf16 W4s[32 * 16];    // K pad 16->32, N pad 8->16
  __shared__ __bf16 W5s[32 * 16];    // K pad 8->32,  N pad 2->16
  __shared__ float  B1s[64], B2s[32], B3s[16], B4s[16], B5s[16];
  // TDM f32 staging for mw1 overlaps the per-wave A tiles: staging is dead
  // (converted to W1s) before the gather phase writes atile. 40 KB.
  __shared__ union {
    float  w1stage[138 * 64];        // 35328 B
    __bf16 atile[8][16 * 160];       // 40960 B
  } U;

  int tid = threadIdx.x;

#if defined(HAVE_TDM)
  if (tid == 0) {
    unsigned long long ga = (unsigned long long)(const void*)mw1;
    unsigned lds_addr = (unsigned)(unsigned long long)(void*)&U.w1stage[0];
    u32x4 g0;
    g0[0] = 1u;                                   // count=1 (valid user D#)
    g0[1] = lds_addr;                             // lds_addr
    g0[2] = (unsigned)(ga & 0xFFFFFFFFu);         // global_addr[31:0]
    g0[3] = (unsigned)((ga >> 32) & 0x01FFFFFFu)  // global_addr[56:32]
            | 0x80000000u;                        // type=2 ("image") in [127:126]
    i32x8 g1;
    g1[0] = 0x20000;        // workgroup_mask=0, data_size=2 (4 bytes)
    g1[1] = (int)(64u << 16);   // tensor_dim0 = 64 (cols, contiguous)
    g1[2] = (int)(138u << 16);  // tensor_dim1 = 138 (rows)
    g1[3] = (int)(64u << 16);   // tile_dim0 = 64
    g1[4] = 138;                // tile_dim1 = 138, tile_dim2 = 0
    g1[5] = 64;                 // tensor_dim0_stride = 64
    g1[6] = 0;
    g1[7] = 0;
    i32x4 gz = {0, 0, 0, 0};
#if defined(TDM_SIX_ARGS)
    i32x8 gz8 = {0, 0, 0, 0, 0, 0, 0, 0};
    __builtin_amdgcn_tensor_load_to_lds(g0, g1, gz, gz, gz8, 0);
#else
    __builtin_amdgcn_tensor_load_to_lds(g0, g1, gz, gz, 0);
#endif
    __builtin_amdgcn_s_wait_tensorcnt(0);         // TENSORcnt -> 0 before barrier
  }
#else
  for (int idx = tid; idx < 138 * 64; idx += 256) U.w1stage[idx] = mw1[idx];
#endif

  // small weights: plain loads while the TDM transfer is in flight
  for (int idx = tid; idx < 64 * 32; idx += 256) W2s[idx] = f2bf(mw2[idx]);
  for (int idx = tid; idx < 32 * 16; idx += 256) W3s[idx] = f2bf(mw3[idx]);
  for (int idx = tid; idx < 32 * 16; idx += 256) {
    int k = idx / 16, nn = idx % 16;
    W4s[idx] = f2bf((k < 16 && nn < 8) ? mw4[k * 8 + nn] : 0.f);
  }
  for (int idx = tid; idx < 32 * 16; idx += 256) {
    int k = idx / 16, nn = idx % 16;
    W5s[idx] = f2bf((k < 8 && nn < 2) ? mw5[k * 2 + nn] : 0.f);
  }
  if (tid < 64) B1s[tid] = mb1[tid];
  if (tid < 32) B2s[tid] = mb2[tid];
  if (tid < 16) B3s[tid] = mb3[tid];
  if (tid < 16) B4s[tid] = (tid < 8) ? mb4[tid] : 0.f;
  if (tid < 16) B5s[tid] = (tid < 2) ? mb5[tid] : 0.f;
  __syncthreads();   // TDM data visible (issuing wave waited TENSORcnt==0)

  // convert staged f32 mw1 -> padded bf16 W1s
  for (int idx = tid; idx < 160 * 64; idx += 256) {
    int k = idx / 64, nn = idx % 64;
    W1s[idx] = f2bf(k < 138 ? U.w1stage[k * 64 + nn] : 0.f);
  }
  __syncthreads();   // staging dead; atile may now be written

  int wave = tid >> 5, lane = tid & 31;
  int m = lane & 15, kgrp = lane >> 4;
  int n = lane & 15;
  int eBase = (blockIdx.x * 8 + wave) * 16;
  __bf16* At = &U.atile[wave][0];

  // ---- gather: lane fills half of row m (cols kgrp*80 .. kgrp*80+79) ----
  {
    int e = eBase + m;
    int s = ei[e], d = ei[E + e];
    const float* xs = x3 + (size_t)s * 64;
    const float* xd = x3 + (size_t)d * 64;
    const float* ev = en + (size_t)e * 10;
    __builtin_prefetch(xs, 0, 0);   // -> global_prefetch_b8
    __builtin_prefetch(xd, 0, 0);
    __bf16* arow = At + m * 160;
    int c0 = kgrp * 80;
#pragma unroll
    for (int j = 0; j < 80; ++j) {
      int c = c0 + j;
      float v = (c < 64) ? xs[c] : (c < 128) ? xd[c - 64] : (c < 138) ? ev[c - 128] : 0.f;
      arow[c] = f2bf(v);
    }
  }
  // per-wave tile: DS ops from same wave are in-order; no block barrier needed.

  // fragment loaders (LDS)
  auto loadA = [&](const __bf16* base, int stride, int kOff) -> v16bf {
    const __bf16* p = base + m * stride + kOff + kgrp * 8;
    v8bf lo = *(const v8bf*)p;          // 16B LDS read
    v8bf hi = *(const v8bf*)(p + 16);   // 16B LDS read
    v16bf r;
#pragma unroll
    for (int i = 0; i < 8; ++i) { r[i] = lo[i]; r[i + 8] = hi[i]; }
    return r;
  };
  auto loadB = [&](const __bf16* Ws, int Ncols, int kOff, int nTile) -> v16bf {
    v16bf b;
#pragma unroll
    for (int j = 0; j < 16; ++j) {
      int k = kOff + ((j < 8) ? kgrp * 8 + j : 16 + kgrp * 8 + (j - 8));
      b[j] = Ws[(size_t)k * Ncols + nTile * 16 + n];
    }
    return b;
  };

  // ---- layer 1: [16x160(138)] @ [160x64] ----
  v8f c1[4];
#pragma unroll
  for (int nt = 0; nt < 4; ++nt) c1[nt] = (v8f){};
#pragma unroll
  for (int kc = 0; kc < 5; ++kc) {
    v16bf a = loadA(At, 160, kc * 32);
#pragma unroll
    for (int nt = 0; nt < 4; ++nt) {
      v16bf b = loadB(W1s, 64, kc * 32, nt);
      c1[nt] = wmma_bf16(a, b, c1[nt]);
    }
  }
#pragma unroll
  for (int nt = 0; nt < 4; ++nt) {
    float bias = B1s[nt * 16 + n];
#pragma unroll
    for (int r = 0; r < 8; ++r) {
      int row = r + 8 * kgrp;
      At[row * 64 + nt * 16 + n] = f2bf(leakyf(c1[nt][r] + bias, 0.1f));
    }
  }

  // ---- layer 2: [16x64] @ [64x32] ----
  v8f c2[2];
#pragma unroll
  for (int nt = 0; nt < 2; ++nt) c2[nt] = (v8f){};
#pragma unroll
  for (int kc = 0; kc < 2; ++kc) {
    v16bf a = loadA(At, 64, kc * 32);
#pragma unroll
    for (int nt = 0; nt < 2; ++nt) {
      v16bf b = loadB(W2s, 32, kc * 32, nt);
      c2[nt] = wmma_bf16(a, b, c2[nt]);
    }
  }
#pragma unroll
  for (int nt = 0; nt < 2; ++nt) {
    float bias = B2s[nt * 16 + n];
#pragma unroll
    for (int r = 0; r < 8; ++r) {
      int row = r + 8 * kgrp;
      At[row * 32 + nt * 16 + n] = f2bf(leakyf(c2[nt][r] + bias, 0.1f));
    }
  }

  // ---- layer 3: [16x32] @ [32x16] ----
  v8f c3 = {};
  {
    v16bf a = loadA(At, 32, 0);
    v16bf b = loadB(W3s, 16, 0, 0);
    c3 = wmma_bf16(a, b, c3);
  }
  {
    float bias = B3s[n];
#pragma unroll
    for (int r = 0; r < 8; ++r) {
      int row = r + 8 * kgrp;
      At[row * 32 + n]      = f2bf(leakyf(c3[r] + bias, 0.1f));
      At[row * 32 + 16 + n] = f2bf(0.f);   // zero-pad K 16->32 for next layer
    }
  }

  // ---- layer 4: [16x32(16)] @ [32x16] (cols 8..15 of W zero) ----
  v8f c4 = {};
  {
    v16bf a = loadA(At, 32, 0);
    v16bf b = loadB(W4s, 16, 0, 0);
    c4 = wmma_bf16(a, b, c4);
  }
  {
    float bias = B4s[n];
#pragma unroll
    for (int r = 0; r < 8; ++r) {
      int row = r + 8 * kgrp;
      At[row * 32 + n]      = f2bf(leakyf(c4[r] + bias, 0.1f));
      At[row * 32 + 16 + n] = f2bf(0.f);
    }
  }

  // ---- layer 5: [16x32(8)] @ [32x16] -> logits in cols 0..1 ----
  v8f c5 = {};
  {
    v16bf a = loadA(At, 32, 0);
    v16bf b = loadB(W5s, 16, 0, 0);
    c5 = wmma_bf16(a, b, c5);
  }
  if (n < 2) {
    float bias = B5s[n];
#pragma unroll
    for (int r = 0; r < 8; ++r) {
      int row = r + 8 * kgrp;
      out[(size_t)(eBase + row) * 2 + n] = c5[r] + bias;
    }
  }
}

// ---------------------------------------------------------------------------
// Host-side orchestration
// ---------------------------------------------------------------------------
static void run_gat(hipStream_t stream, const float* xin, int K, int outdim,
                    const float* W, const float* a_s, const float* a_d, const float* bias,
                    const int* ei, float* h, float* asrc, float* adst, int* amax,
                    float* denom, float* al, float* ea, float* agg, float* xout) {
  const int N = NNODES, E = NEDGES, Etot = NEDGES + NNODES, H = GAT_HEADS;
  const int Ncols = H * outdim;
  int tiles = (N / 16) * (Ncols / 16);
  gemm16_bf16_kernel<<<(tiles + 7) / 8, 256, 0, stream>>>(xin, W, h, N, K, Ncols);
  int nh = N * H;
  attn_score_kernel<<<(nh + 255) / 256, 256, 0, stream>>>(h, a_s, a_d, asrc, adst, N, outdim);
  init_amax_kernel<<<(nh + 255) / 256, 256, 0, stream>>>(amax, nh);
  hipMemsetAsync(denom, 0, (size_t)nh * sizeof(float), stream);
  hipMemsetAsync(agg, 0, (size_t)nh * outdim * sizeof(float), stream);
  int eh = Etot * H;
  edge_score_kernel<<<(eh + 255) / 256, 256, 0, stream>>>(ei, E, Etot, asrc, adst, al, amax);
  edge_exp_kernel<<<(eh + 255) / 256, 256, 0, stream>>>(ei, E, Etot, al, amax, ea, denom);
  edge_agg_kernel<<<(eh + 255) / 256, 256, 0, stream>>>(ei, E, Etot, ea, denom, h, agg, outdim);
  int nd = N * outdim;
  head_mean_kernel<<<(nd + 255) / 256, 256, 0, stream>>>(agg, bias, xout, N, outdim);
}

extern "C" void kernel_launch(void* const* d_in, const int* in_sizes, int n_in,
                              void* d_out, int out_size, void* d_ws, size_t ws_size,
                              hipStream_t stream) {
  const int N = NNODES, E = NEDGES, H = GAT_HEADS;
  const int Etot = E + N;

  const float* x   = (const float*)d_in[0];
  const int*   ei  = (const int*)d_in[1];
  const float* e   = (const float*)d_in[2];
  const float* bng = (const float*)d_in[4];
  const float* bnb = (const float*)d_in[5];
  const float* beg = (const float*)d_in[6];
  const float* beb = (const float*)d_in[7];
  const float* W1  = (const float*)d_in[8];
  const float* as1 = (const float*)d_in[9];
  const float* ad1 = (const float*)d_in[10];
  const float* b1  = (const float*)d_in[11];
  const float* W2  = (const float*)d_in[12];
  const float* as2 = (const float*)d_in[13];
  const float* ad2 = (const float*)d_in[14];
  const float* b2  = (const float*)d_in[15];
  const float* W3  = (const float*)d_in[16];
  const float* as3 = (const float*)d_in[17];
  const float* ad3 = (const float*)d_in[18];
  const float* b3  = (const float*)d_in[19];
  const float* mw1 = (const float*)d_in[20];  const float* mb1 = (const float*)d_in[21];
  const float* mw2 = (const float*)d_in[22];  const float* mb2 = (const float*)d_in[23];
  const float* mw3 = (const float*)d_in[24];  const float* mb3 = (const float*)d_in[25];
  const float* mw4 = (const float*)d_in[26];  const float* mb4 = (const float*)d_in[27];
  const float* mw5 = (const float*)d_in[28];  const float* mb5 = (const float*)d_in[29];
  float* out = (float*)d_out;

  // workspace layout (float units)
  float* ws = (float*)d_ws;
  size_t off = 0;
  float* xn    = ws + off; off += (size_t)N * 16;          // 800k
  float* en    = ws + off; off += (size_t)E * 10;          // 8M
  float* stats = ws + off; off += 64;                      // sx[16] sqx[16] se[16] sqe[16]
  float* h     = ws + off; off += (size_t)N * H * 64;      // 9.6M (max layer)
  float* asrc  = ws + off; off += (size_t)N * H;
  float* adst  = ws + off; off += (size_t)N * H;
  int*   amax  = (int*)(ws + off); off += (size_t)N * H;
  float* denom = ws + off; off += (size_t)N * H;
  float* al    = ws + off; off += (size_t)Etot * H;
  float* ea    = ws + off; off += (size_t)Etot * H;
  float* agg   = ws + off; off += (size_t)N * H * 64;
  float* x1    = ws + off; off += (size_t)N * 16;
  float* x2    = ws + off; off += (size_t)N * 32;
  float* x3    = ws + off; off += (size_t)N * 64;
  (void)in_sizes; (void)n_in; (void)out_size; (void)ws_size;

  // --- BatchNorm on nodes and edges ---
  hipMemsetAsync(stats, 0, 64 * sizeof(float), stream);
  colstats_kernel<<<256, 256, 0, stream>>>(x, N, 16, stats + 0, stats + 16);
  colstats_kernel<<<512, 256, 0, stream>>>(e, E, 10, stats + 32, stats + 48);
  bn_apply_kernel<<<(N * 16 + 255) / 256, 256, 0, stream>>>(x, xn, stats + 0, stats + 16, bng, bnb, N, 16);
  bn_apply_kernel<<<(E * 10 + 255) / 256, 256, 0, stream>>>(e, en, stats + 32, stats + 48, beg, beb, E, 10);

  // --- GAT stack ---
  run_gat(stream, xn, 16, 16, W1, as1, ad1, b1, ei, h, asrc, adst, amax, denom, al, ea, agg, x1);
  run_gat(stream, x1, 16, 32, W2, as2, ad2, b2, ei, h, asrc, adst, amax, denom, al, ea, agg, x2);
  run_gat(stream, x2, 32, 64, W3, as3, ad3, b3, ei, h, asrc, adst, amax, denom, al, ea, agg, x3);

  // --- fused edge MLP (all five layers chained through LDS + WMMA) ---
  edge_mlp_kernel<<<E / 128, 256, 0, stream>>>(x3, ei, en,
                                               mw1, mb1, mw2, mb2, mw3, mb3, mw4, mb4, mw5, mb5,
                                               out, E);
}
